// CustomMamba_63780264345601
// MI455X (gfx1250) — matrix operand order repaired and checked by reading, not compiled, not run on hardware
//
#include <hip/hip_runtime.h>
#include <hip/hip_bf16.h>

typedef __attribute__((ext_vector_type(16))) _Float16 v16h_t;
typedef __attribute__((ext_vector_type(8)))  _Float16 v8h_t;
typedef __attribute__((ext_vector_type(8)))  float    v8f_t;

#define DEV __device__ __forceinline__

DEV v8f_t wmma16(v16h_t a, v16h_t b, v8f_t c) {
  // D = A(16x32 f16) * B(32x16 f16) + C(16x16 f32)
  return __builtin_amdgcn_wmma_f32_16x16x32_f16(false, a, false, b, (short)0, c,
                                                false, false);
}

DEV float sigmoidf_(float x) { return 1.f / (1.f + __expf(-x)); }
DEV float siluf_(float x)    { return x / (1.f + __expf(-x)); }
DEV float softplusf_(float x){ return x > 20.f ? x : log1pf(__expf(x)); }

// ---------------------------------------------------------------------------
// Weight convert+transpose: src fp32 [K x N] row-major -> dst f16 [Npad x Kpad]
// (N-major so WMMA B fragments load contiguously); pad region zero-filled.
// ---------------------------------------------------------------------------
__global__ void wconv_kernel(const float* __restrict__ src, _Float16* __restrict__ dst,
                             int K, int N, int Kpad, int Npad) {
  int idx = blockIdx.x * 256 + threadIdx.x;
  int total = Kpad * Npad;
  if (idx >= total) return;
  int n = idx / Kpad;
  int k = idx - n * Kpad;
  float v = (n < N && k < K) ? src[(size_t)k * N + n] : 0.f;
  dst[idx] = (_Float16)v;
}

// ---------------------------------------------------------------------------
// Embedding: h[row,hh] = emb[ids[row],hh] + times[row]*time_w[hh] + time_b[hh]
// ---------------------------------------------------------------------------
__global__ void embed_kernel(const int* __restrict__ ids, const float* __restrict__ times,
                             const float* __restrict__ emb, const float* __restrict__ tw,
                             const float* __restrict__ tb, float* __restrict__ h,
                             int H, int total) {
  int idx = blockIdx.x * 256 + threadIdx.x;
  if (idx >= total) return;
  int rid = idx / H;
  int hh  = idx - rid * H;
  h[idx] = emb[(size_t)ids[rid] * H + hh] + times[rid] * tw[hh] + tb[hh];
}

// ---------------------------------------------------------------------------
// RMSNorm over H=768 per row, scaled by w, written as f16 (WMMA A operand).
// ---------------------------------------------------------------------------
__global__ __launch_bounds__(256)
void rmsnorm_f16_kernel(const float* __restrict__ h, const float* __restrict__ w,
                        _Float16* __restrict__ out, int H) {
  __shared__ float red[256];
  const int row = blockIdx.x;
  const float* hr = h + (size_t)row * H;
  float s = 0.f;
  for (int i = threadIdx.x; i < H; i += 256) { float v = hr[i]; s += v * v; }
  red[threadIdx.x] = s;
  __syncthreads();
  for (int st = 128; st > 0; st >>= 1) {
    if ((int)threadIdx.x < st) red[threadIdx.x] += red[threadIdx.x + st];
    __syncthreads();
  }
  float inv = rsqrtf(red[0] / (float)H + 1e-5f);
  for (int i = threadIdx.x; i < H; i += 256)
    out[(size_t)row * H + i] = (_Float16)(hr[i] * inv * w[i]);
}

// ---------------------------------------------------------------------------
// Generic WMMA GEMM: Out[M,N](f32 or f16) = act( A[M,Kp]f16 * Bt[N,Kp]f16^T
//                                               + bias[n] + Cin[M,ldc] )
// One wave computes a 32x32 tile with 4 f32 accumulators.
// Fragment layouts used here (A and B share contiguous K order, not the ISA table order):
//   A 16x32 f16: lane(0-15)=row r, K in [0,16); lane(16-31): K in [16,32)
//   B 32x16 f16: lane holds col n=lane&15, 16 contiguous K at (lane>>4)*16
//   C/D 16x16 f32: vgpr v -> row v + 8*(lane>>4), col lane&15
// ---------------------------------------------------------------------------
__global__ __launch_bounds__(128)
void gemm_wmma_kernel(const _Float16* __restrict__ A, const _Float16* __restrict__ Bt,
                      const float* __restrict__ Cin, void* __restrict__ Out,
                      int M, int N, int Kp, int ldc,
                      const float* __restrict__ bias, int act, int out_half) {
  const int lane = threadIdx.x & 31;
  const int wave = threadIdx.x >> 5;
  const int tiles_n = N >> 5;
  const int total   = (M >> 5) * tiles_n;
  int tile = blockIdx.x * 4 + wave;
  if (tile >= total) return;                 // wave-uniform exit, EXEC stays full
  const int tm = tile / tiles_n;
  const int tn = tile - tm * tiles_n;
  const int m0 = tm << 5, n0 = tn << 5;
  const int r  = lane & 15;
  const int hs = lane >> 4;

  const _Float16* ap = A  + (size_t)(m0 + r) * Kp + hs * 16;
  const _Float16* bp = Bt + (size_t)(n0 + r) * Kp + hs * 16;
  const size_t step16 = (size_t)16 * Kp;

  v8f_t c00 = {}, c01 = {}, c10 = {}, c11 = {};
  for (int k = 0; k < Kp; k += 32) {
    v8h_t a0lo = *(const v8h_t*)(ap + k);
    v8h_t a0hi = *(const v8h_t*)(ap + k + 8);
    v8h_t a1lo = *(const v8h_t*)(ap + step16 + k);
    v8h_t a1hi = *(const v8h_t*)(ap + step16 + k + 8);
    v16h_t b0  = *(const v16h_t*)(bp + k);
    v16h_t b1  = *(const v16h_t*)(bp + step16 + k);
    v16h_t a0 = __builtin_shufflevector(a0lo, a0hi, 0,1,2,3,4,5,6,7,8,9,10,11,12,13,14,15);
    v16h_t a1 = __builtin_shufflevector(a1lo, a1hi, 0,1,2,3,4,5,6,7,8,9,10,11,12,13,14,15);
    c00 = wmma16(a0, b0, c00);
    c01 = wmma16(a0, b1, c01);
    c10 = wmma16(a1, b0, c10);
    c11 = wmma16(a1, b1, c11);
  }

  const int cn = n0 + r;
  const float bv0 = bias ? bias[cn] : 0.f;
  const float bv1 = bias ? bias[cn + 16] : 0.f;
#pragma unroll
  for (int v = 0; v < 8; ++v) {
    const int row0 = m0 + v + hs * 8;
    const int row1 = row0 + 16;
    float x00 = c00[v] + bv0;
    float x01 = c01[v] + bv1;
    float x10 = c10[v] + bv0;
    float x11 = c11[v] + bv1;
    if (Cin) {
      x00 += Cin[(size_t)row0 * ldc + cn];
      x01 += Cin[(size_t)row0 * ldc + cn + 16];
      x10 += Cin[(size_t)row1 * ldc + cn];
      x11 += Cin[(size_t)row1 * ldc + cn + 16];
    }
    if (act == 1) {
      x00 = softplusf_(x00); x01 = softplusf_(x01);
      x10 = softplusf_(x10); x11 = softplusf_(x11);
    }
    if (out_half) {
      _Float16* o = (_Float16*)Out;
      o[(size_t)row0 * ldc + cn]      = (_Float16)x00;
      o[(size_t)row0 * ldc + cn + 16] = (_Float16)x01;
      o[(size_t)row1 * ldc + cn]      = (_Float16)x10;
      o[(size_t)row1 * ldc + cn + 16] = (_Float16)x11;
    } else {
      float* o = (float*)Out;
      o[(size_t)row0 * ldc + cn]      = x00;
      o[(size_t)row0 * ldc + cn + 16] = x01;
      o[(size_t)row1 * ldc + cn]      = x10;
      o[(size_t)row1 * ldc + cn + 16] = x11;
    }
  }
}

// ---------------------------------------------------------------------------
// Causal depthwise conv (K=4) over xs = xz[:, 0:I], + bias, SiLU.
// Writes f32 (for scan) and f16 (GEMM A for x_proj).
// ---------------------------------------------------------------------------
__global__ void conv_silu_kernel(const _Float16* __restrict__ xz, const float* __restrict__ cw,
                                 const float* __restrict__ cb, float* __restrict__ xc32,
                                 _Float16* __restrict__ xc16, int S, int I, int W2, int total) {
  int idx = blockIdx.x * 256 + threadIdx.x;
  if (idx >= total) return;
  int i   = idx % I;
  int rid = idx / I;
  int s   = rid & (S - 1);
  float acc = cb[i];
#pragma unroll
  for (int t = 0; t < 4; ++t) {
    int sp = s - 3 + t;
    if (sp >= 0) acc += (float)xz[(size_t)(rid - 3 + t) * W2 + i] * cw[i * 4 + t];
  }
  float v = siluf_(acc);
  xc32[idx] = v;
  xc16[idx] = (_Float16)v;
}

// dtr columns 0..47 of proj -> f16, K padded to 64 with zeros.
__global__ void dtr_pack_kernel(const float* __restrict__ proj, _Float16* __restrict__ dtr,
                                int R, int Rpad, int Pw, int total) {
  int idx = blockIdx.x * 256 + threadIdx.x;
  if (idx >= total) return;
  int row = idx / Rpad;
  int j   = idx - row * Rpad;
  dtr[idx] = (_Float16)(j < R ? proj[(size_t)row * Pw + j] : 0.f);
}

// ---------------------------------------------------------------------------
// Selective scan: one wave32 per (batch, 32 channels). hs[16] live in VGPRs.
// Per-step B_t/C_t (32 floats, shared by all channels) read one-per-lane and
// broadcast via wave shuffles. exp via hardware transcendental.
// ---------------------------------------------------------------------------
__global__ __launch_bounds__(32)
void scan_kernel(const float* __restrict__ dtv, const float* __restrict__ xc32,
                 const float* __restrict__ proj, const float* __restrict__ Alog,
                 float* __restrict__ ys, int S, int I, int Pw, int blocksPerB) {
  const int lane = threadIdx.x;
  const int b  = blockIdx.x / blocksPerB;
  const int ic = (blockIdx.x % blocksPerB) * 32 + lane;
  float a[16];
#pragma unroll
  for (int n = 0; n < 16; ++n) a[n] = -__expf(Alog[(size_t)ic * 16 + n]);
  float hstate[16];
#pragma unroll
  for (int n = 0; n < 16; ++n) hstate[n] = 0.f;

  const size_t base = (size_t)b * S;
  for (int s = 0; s < S; ++s) {
    const size_t row = base + s;
    float dt  = dtv[row * I + ic];
    float xv  = xc32[row * I + ic];
    float pv  = proj[row * Pw + 48 + lane];   // lanes 0-15: B_t, 16-31: C_t
    float dtx = dt * xv;
    float y = 0.f;
#pragma unroll
    for (int n = 0; n < 16; ++n) {
      float Bn = __shfl(pv, n, 32);
      float Cn = __shfl(pv, 16 + n, 32);
      float e  = __expf(dt * a[n]);
      hstate[n] = hstate[n] * e + dtx * Bn;
      y += hstate[n] * Cn;
    }
    ys[row * I + ic] = y;
  }
}

// y16 = (ys + xc*D) * silu(z),  z = xz[:, I + i]
__global__ void gate_kernel(const float* __restrict__ ys, const float* __restrict__ xc32,
                            const _Float16* __restrict__ xz, const float* __restrict__ Dv,
                            _Float16* __restrict__ y16, int I, int W2, int total) {
  int idx = blockIdx.x * 256 + threadIdx.x;
  if (idx >= total) return;
  int i   = idx % I;
  int rid = idx / I;
  float z = (float)xz[(size_t)rid * W2 + I + i];
  float y = (ys[idx] + xc32[idx] * Dv[i]) * siluf_(z);
  y16[idx] = (_Float16)y;
}

// masked mean-pool over S of rmsnorm'ed hidden (already in f16 buffer xn)
__global__ void pool_kernel(const _Float16* __restrict__ xn, const float* __restrict__ mask,
                            float* __restrict__ pooled, int S, int H, int total) {
  int idx = blockIdx.x * 256 + threadIdx.x;
  if (idx >= total) return;
  int b  = idx / H;
  int hh = idx - b * H;
  float acc = 0.f, dm = 0.f;
  for (int s = 0; s < S; ++s) {
    float m = mask[b * S + s];
    acc += m * (float)xn[(size_t)(b * S + s) * H + hh];
    dm  += m;
  }
  pooled[idx] = acc / fmaxf(dm, 1e-9f);
}

// classifier: out[b] = relu(pooled @ w1 + b1) @ w2 + b2   (768 -> 384 -> 1)
__global__ __launch_bounds__(384)
void cls_kernel(const float* __restrict__ pooled, const float* __restrict__ w1,
                const float* __restrict__ b1, const float* __restrict__ w2,
                const float* __restrict__ b2, float* __restrict__ out, int H, int Hh) {
  __shared__ float red[384];
  const int b = blockIdx.x, j = threadIdx.x;
  const float* p = pooled + (size_t)b * H;
  float a = b1[j];
  for (int k = 0; k < H; ++k) a += p[k] * w1[(size_t)k * Hh + j];
  a = fmaxf(a, 0.f);
  red[j] = a * w2[j];
  __syncthreads();
  if (j == 0) {
    float s = 0.f;
    for (int k = 0; k < 384; ++k) s += red[k];
    out[b] = s + b2[0];
  }
}

// ---------------------------------------------------------------------------
extern "C" void kernel_launch(void* const* d_in, const int* in_sizes, int n_in,
                              void* d_out, int out_size, void* d_ws, size_t ws_size,
                              hipStream_t stream) {
  (void)in_sizes; (void)n_in; (void)out_size; (void)ws_size;
  const int H = 768, I = 1536, R = 48, L = 24, B = 8, S = 1024;
  const int M  = B * S;       // 8192 rows
  const int N2 = 2 * I;       // 3072
  const int Pw = 96;          // proj width padded 80 -> 96
  const int Rp = 64;          // dtr K padded 48 -> 64

  const int*   input_ids = (const int*)d_in[0];
  const float* times     = (const float*)d_in[1];
  const float* mask      = (const float*)d_in[2];
  const float* emb       = (const float*)d_in[3];
  const float* time_w    = (const float*)d_in[4];
  const float* time_b    = (const float*)d_in[5];
  const float* ln_w      = (const float*)d_in[6];
  const float* in_proj_w = (const float*)d_in[7];
  const float* conv_w    = (const float*)d_in[8];
  const float* conv_b    = (const float*)d_in[9];
  const float* x_proj_w  = (const float*)d_in[10];
  const float* dt_w      = (const float*)d_in[11];
  const float* dt_b      = (const float*)d_in[12];
  const float* A_log     = (const float*)d_in[13];
  const float* Dw        = (const float*)d_in[14];
  const float* out_w     = (const float*)d_in[15];
  const float* normf_w   = (const float*)d_in[16];
  const float* cls_w1    = (const float*)d_in[17];
  const float* cls_b1    = (const float*)d_in[18];
  const float* cls_w2    = (const float*)d_in[19];
  const float* cls_b2    = (const float*)d_in[20];

  char* wsb = (char*)d_ws;
  size_t off = 0;
  auto take = [&](size_t bytes) -> void* {
    void* p = wsb + off;
    off += (bytes + 255) & ~(size_t)255;
    return p;
  };
  _Float16* w_in  = (_Float16*)take((size_t)L * N2 * H * 2);
  _Float16* w_x   = (_Float16*)take((size_t)L * Pw * I * 2);
  _Float16* w_dt  = (_Float16*)take((size_t)L * I * Rp * 2);
  _Float16* w_out = (_Float16*)take((size_t)L * H * I * 2);
  float*    hbuf  = (float*)take((size_t)M * H * 4);
  _Float16* x16   = (_Float16*)take((size_t)M * H * 2);
  _Float16* xz16  = (_Float16*)take((size_t)M * N2 * 2);
  float*    xc32  = (float*)take((size_t)M * I * 4);
  _Float16* xc16  = (_Float16*)take((size_t)M * I * 2);
  float*    proj  = (float*)take((size_t)M * Pw * 4);
  _Float16* dtr16 = (_Float16*)take((size_t)M * Rp * 2);
  float*    dtv   = (float*)take((size_t)M * I * 4);
  float*    ysb   = (float*)take((size_t)M * I * 4);
  _Float16* y16   = (_Float16*)take((size_t)M * I * 2);
  float*    pooled= (float*)take((size_t)B * H * 4);

  auto wc = [&](const float* src, _Float16* dst, int K, int N, int Kpad, int Npad) {
    int total = Kpad * Npad;
    wconv_kernel<<<(total + 255) / 256, 256, 0, stream>>>(src, dst, K, N, Kpad, Npad);
  };
  // Convert + transpose all weights to f16 (N-major, padded) once per call.
  for (int l = 0; l < L; ++l) {
    wc(in_proj_w + (size_t)l * H * N2, w_in  + (size_t)l * N2 * H, H, N2, H, N2);
    wc(x_proj_w  + (size_t)l * I * 80, w_x   + (size_t)l * Pw * I, I, 80, I, Pw);
    wc(dt_w      + (size_t)l * R * I,  w_dt  + (size_t)l * I * Rp, R, I, Rp, I);
    wc(out_w     + (size_t)l * I * H,  w_out + (size_t)l * H * I,  I, H, I, H);
  }

  auto gemm = [&](const _Float16* A, const _Float16* Bt, const float* Cin, void* Out,
                  int Mm, int Nn, int Kp, int ldc, const float* bias, int act, int ohalf) {
    int tiles = (Mm / 32) * (Nn / 32);
    gemm_wmma_kernel<<<(tiles + 3) / 4, 128, 0, stream>>>(A, Bt, Cin, Out, Mm, Nn, Kp,
                                                          ldc, bias, act, ohalf);
  };

  embed_kernel<<<(M * H + 255) / 256, 256, 0, stream>>>(input_ids, times, emb, time_w,
                                                        time_b, hbuf, H, M * H);
  for (int l = 0; l < L; ++l) {
    rmsnorm_f16_kernel<<<M, 256, 0, stream>>>(hbuf, ln_w + (size_t)l * H, x16, H);
    // xz = x @ in_proj  (f16 out)
    gemm(x16, w_in + (size_t)l * N2 * H, nullptr, xz16, M, N2, H, N2, nullptr, 0, 1);
    conv_silu_kernel<<<(M * I + 255) / 256, 256, 0, stream>>>(
        xz16, conv_w + (size_t)l * I * 4, conv_b + (size_t)l * I, xc32, xc16, S, I, N2, M * I);
    // proj = xc @ x_proj  (f32 out, padded width 96)
    gemm(xc16, w_x + (size_t)l * Pw * I, nullptr, proj, M, Pw, I, Pw, nullptr, 0, 0);
    dtr_pack_kernel<<<(M * Rp + 255) / 256, 256, 0, stream>>>(proj, dtr16, R, Rp, Pw, M * Rp);
    // dt = softplus(dtr @ dt_w + dt_b)
    gemm(dtr16, w_dt + (size_t)l * I * Rp, nullptr, dtv, M, I, Rp, I,
         dt_b + (size_t)l * I, 1, 0);
    scan_kernel<<<B * (I / 32), 32, 0, stream>>>(dtv, xc32, proj,
                                                 A_log + (size_t)l * I * 16, ysb,
                                                 S, I, Pw, I / 32);
    gate_kernel<<<(M * I + 255) / 256, 256, 0, stream>>>(ysb, xc32, xz16,
                                                         Dw + (size_t)l * I, y16, I, N2, M * I);
    // resid += y @ out_w   (beta=1 via Cin)
    gemm(y16, w_out + (size_t)l * H * I, hbuf, hbuf, M, H, I, H, nullptr, 0, 0);
  }
  // final rmsnorm (scaled by normf_w) -> f16, masked mean pool, classifier
  rmsnorm_f16_kernel<<<M, 256, 0, stream>>>(hbuf, normf_w, x16, H);
  pool_kernel<<<(B * H + 255) / 256, 256, 0, stream>>>(x16, mask, pooled, S, H, B * H);
  cls_kernel<<<B, 384, 0, stream>>>(pooled, cls_w1, cls_b1, cls_w2, cls_b2,
                                    (float*)d_out, H, H / 2);
}
